// DecoderLayer_39994735460721
// MI455X (gfx1250) — matrix operand order repaired and checked by reading, not compiled, not run on hardware
//
#include <hip/hip_runtime.h>
#include <math.h>

typedef __attribute__((ext_vector_type(16))) __bf16 v16bf;
typedef __attribute__((ext_vector_type(8)))  float  v8f;

#define THREADS 256
#define SEQ 2048   // sequence length (compile-time for epilogue head packing)

// ---------------------------------------------------------------------------
// Fragment-layout index helpers (CDNA5 ISA §7.12.2, wave32).
// A (16x32 bf16, MxK): lane = m%16 (+16 for K-subblocks 8..15/24..31),
//   half h: K = (h<8 ? h : h+8) + (lane>=16 ? 8 : 0)
// B (32x16 bf16, KxN): lane = n%16 (+16 for K-subblocks 8..15/24..31),
//   half h: K = (h<8 ? h : h+8) + (lane>=16 ? 8 : 0)
// Storage: [mt|nt][kt][lane][16 halves] -> one 32B contiguous load per lane.
// ---------------------------------------------------------------------------
__device__ __forceinline__ size_t pa_idx(int m, int k, int Kt) {
    int kk   = k & 31;
    int lane = (m & 15) | (((kk >> 3) & 1) << 4);
    int h    = (kk & 7) | (((kk >> 4) & 1) << 3);
    return (((size_t)(m >> 4) * (size_t)Kt + (size_t)(k >> 5)) * 32 + lane) * 16 + h;
}
__device__ __forceinline__ size_t pb_idx(int k, int n, int Kt) {
    int kk   = k & 31;
    int lane = (n & 15) | (((kk >> 3) & 1) << 4);
    int h    = (kk & 7) | (((kk >> 4) & 1) << 3);
    return (((size_t)(n >> 4) * (size_t)Kt + (size_t)(k >> 5)) * 32 + lane) * 16 + h;
}

// Pack fp32 (m,k) = src[m*rs + k]  ->  bf16 A-fragment layout
__global__ __launch_bounds__(THREADS) void k_pack_a(const float* __restrict__ src,
                                                    __bf16* __restrict__ dst,
                                                    int M, int K, int rs) {
    size_t idx = (size_t)blockIdx.x * THREADS + threadIdx.x;
    if (idx >= (size_t)M * K) return;
    int m = (int)(idx / K), k = (int)(idx % K);
    dst[pa_idx(m, k, K >> 5)] = (__bf16)src[(size_t)m * rs + k];
}

// Pack fp32 (k,n) = src[k*rs + n]  ->  bf16 B-fragment layout
__global__ __launch_bounds__(THREADS) void k_pack_b(const float* __restrict__ src,
                                                    __bf16* __restrict__ dst,
                                                    int K, int N, int rs) {
    size_t idx = (size_t)blockIdx.x * THREADS + threadIdx.x;
    if (idx >= (size_t)K * N) return;
    int k = (int)(idx / N), n = (int)(idx % N);
    dst[pb_idx(k, n, K >> 5)] = (__bf16)src[(size_t)k * rs + n];
}

// ---------------------------------------------------------------------------
// Epilogue modes (compile-time; each kernel instantiation carries only one):
//  0: fp32 row-major C (ldc), +bias, +optional relu
//  1: packed-A bf16 at (rowOff+m, colOff+n) with dest K-tiles = packKt
//  2: per-(batch,head) Q pack  (A-fmt, K=64):  buf[b*16+h], pa_idx(s, kl, 2)
//  3: per-(batch,head) K^T pack(B-fmt, K=64):  buf[b*16+h], pb_idx(kl, s, 2)
//  4: per-(batch,head) V pack  (B-fmt, K=SEQ): buf[b*16+h], pb_idx(s, nl, SEQ/32)
// ---------------------------------------------------------------------------
template <int MODE>
__device__ __forceinline__ void epi_store(const v8f& acc,
                                          float* __restrict__ C, int ldc,
                                          __bf16* __restrict__ pd, int packKt,
                                          int rowOff, int colOff,
                                          const float* __restrict__ bias,
                                          int relu, int mt, int nt, int lane) {
    int col   = nt * 16 + (lane & 15);
    int rbase = mt * 16 + ((lane >> 4) << 3);   // lanes 16-31 hold rows M=8..15
    float bv  = bias ? bias[col] : 0.0f;
#pragma unroll
    for (int r = 0; r < 8; ++r) {
        float v = acc[r] + bv;
        if (relu) v = fmaxf(v, 0.0f);
        int m = rbase + r;
        if constexpr (MODE == 0) {
            C[(size_t)m * ldc + col] = v;
        } else if constexpr (MODE == 1) {
            pd[pa_idx(rowOff + m, colOff + col, packKt)] = (__bf16)v;
        } else {
            int b = m >> 11, s = m & (SEQ - 1);
            int head = col >> 6, kl = col & 63;
            __bf16* buf = pd + ((size_t)((b << 4) + head)) * (size_t)SEQ * 64;
            if constexpr (MODE == 2)      buf[pa_idx(s, kl, 2)] = (__bf16)v;
            else if constexpr (MODE == 3) buf[pb_idx(kl, s, 2)] = (__bf16)v;
            else                          buf[pb_idx(s, kl, SEQ >> 5)] = (__bf16)v;
        }
    }
}

// ---------------------------------------------------------------------------
// Big GEMM: 8 waves, wave tile 2Mx4N 16x16 tiles (32x64), block 128x128.
// Per k-step: 6 fragment loads feed 8 v_wmma_f32_16x16x32_bf16.
// Requires M%128==0, N%128==0.
// ---------------------------------------------------------------------------
template <int MODE>
__global__ __launch_bounds__(THREADS) void k_gemm_big(const __bf16* __restrict__ Ap,
                                                      const __bf16* __restrict__ Bp,
                                                      int Kt,
                                                      float* __restrict__ C, int ldc,
                                                      __bf16* __restrict__ pd, int packKt,
                                                      int rowOff, int colOff,
                                                      const float* __restrict__ bias,
                                                      int relu) {
    int lane = threadIdx.x & 31;
    int w    = threadIdx.x >> 5;
    int mt0  = blockIdx.x * 8 + (w >> 1) * 2;   // 4 waves in M, 2 tiles each
    int nt0  = blockIdx.y * 8 + (w & 1) * 4;    // 2 waves in N, 4 tiles each

    const __bf16* ap = Ap + ((size_t)mt0 * Kt) * 512 + (size_t)lane * 16;
    const __bf16* bp = Bp + ((size_t)nt0 * Kt) * 512 + (size_t)lane * 16;
    const size_t astr = (size_t)Kt * 512;

    v8f acc[2][4];
#pragma unroll
    for (int i = 0; i < 2; ++i)
#pragma unroll
        for (int j = 0; j < 4; ++j) acc[i][j] = (v8f){};

    for (int kt = 0; kt < Kt; ++kt) {
        v16bf a[2], b[4];
#pragma unroll
        for (int i = 0; i < 2; ++i)
            a[i] = *(const v16bf*)(ap + (size_t)i * astr + (size_t)kt * 512);
#pragma unroll
        for (int j = 0; j < 4; ++j)
            b[j] = *(const v16bf*)(bp + (size_t)j * astr + (size_t)kt * 512);
#pragma unroll
        for (int i = 0; i < 2; ++i)
#pragma unroll
            for (int j = 0; j < 4; ++j)
                acc[i][j] = __builtin_amdgcn_wmma_f32_16x16x32_bf16(
                    false, a[i], false, b[j], (short)0, acc[i][j], false, false);
    }
#pragma unroll
    for (int i = 0; i < 2; ++i)
#pragma unroll
        for (int j = 0; j < 4; ++j)
            epi_store<MODE>(acc[i][j], C, ldc, pd, packKt, rowOff, colOff,
                            bias, relu, mt0 + i, nt0 + j, lane);
}

// ---------------------------------------------------------------------------
// Small GEMM: 2Mx2N wave tile, block 128x64 (for narrow N, e.g. P@V with N=64).
// ---------------------------------------------------------------------------
template <int MODE>
__global__ __launch_bounds__(THREADS) void k_gemm_small(const __bf16* __restrict__ Ap,
                                                        const __bf16* __restrict__ Bp,
                                                        int Kt,
                                                        float* __restrict__ C, int ldc,
                                                        __bf16* __restrict__ pd, int packKt,
                                                        int rowOff, int colOff,
                                                        const float* __restrict__ bias,
                                                        int relu) {
    int lane = threadIdx.x & 31;
    int w    = threadIdx.x >> 5;
    int mt0  = blockIdx.x * 8 + (w >> 1) * 2;
    int nt0  = blockIdx.y * 4 + (w & 1) * 2;

    const __bf16* ap = Ap + ((size_t)mt0 * Kt) * 512 + (size_t)lane * 16;
    const __bf16* bp = Bp + ((size_t)nt0 * Kt) * 512 + (size_t)lane * 16;
    const size_t astr = (size_t)Kt * 512;

    v8f acc[2][2];
#pragma unroll
    for (int i = 0; i < 2; ++i)
#pragma unroll
        for (int j = 0; j < 2; ++j) acc[i][j] = (v8f){};

    for (int kt = 0; kt < Kt; ++kt) {
        v16bf a[2], b[2];
#pragma unroll
        for (int i = 0; i < 2; ++i)
            a[i] = *(const v16bf*)(ap + (size_t)i * astr + (size_t)kt * 512);
#pragma unroll
        for (int j = 0; j < 2; ++j)
            b[j] = *(const v16bf*)(bp + (size_t)j * astr + (size_t)kt * 512);
#pragma unroll
        for (int i = 0; i < 2; ++i)
#pragma unroll
            for (int j = 0; j < 2; ++j)
                acc[i][j] = __builtin_amdgcn_wmma_f32_16x16x32_bf16(
                    false, a[i], false, b[j], (short)0, acc[i][j], false, false);
    }
#pragma unroll
    for (int i = 0; i < 2; ++i)
#pragma unroll
        for (int j = 0; j < 2; ++j)
            epi_store<MODE>(acc[i][j], C, ldc, pd, packKt, rowOff, colOff,
                            bias, relu, mt0 + i, nt0 + j, lane);
}

// ---------------------------------------------------------------------------
// Causal softmax of one score row -> probs directly in packed-A bf16 layout
// (zeros above the diagonal so downstream WMMA consumes full K).
// ---------------------------------------------------------------------------
__global__ __launch_bounds__(THREADS) void k_softmax_pack(const float* __restrict__ sc,
                                                          __bf16* __restrict__ dst,
                                                          int Scols, float scale) {
    int s = blockIdx.x;
    const float* row = sc + (size_t)s * Scols;
    int L = s + 1;
    __shared__ float red[THREADS];
    int tid = threadIdx.x;

    float mx = -INFINITY;
    for (int k = tid; k < L; k += THREADS) mx = fmaxf(mx, row[k] * scale);
    red[tid] = mx; __syncthreads();
    for (int o = THREADS / 2; o > 0; o >>= 1) {
        if (tid < o) red[tid] = fmaxf(red[tid], red[tid + o]);
        __syncthreads();
    }
    mx = red[0]; __syncthreads();

    float sum = 0.0f;
    for (int k = tid; k < L; k += THREADS) sum += __expf(row[k] * scale - mx);
    red[tid] = sum; __syncthreads();
    for (int o = THREADS / 2; o > 0; o >>= 1) {
        if (tid < o) red[tid] += red[tid + o];
        __syncthreads();
    }
    float inv = 1.0f / red[0];

    int Kt = Scols >> 5;
    for (int k = tid; k < Scols; k += THREADS) {
        float p = (k < L) ? __expf(row[k] * scale - mx) * inv : 0.0f;
        dst[pa_idx(s, k, Kt)] = (__bf16)p;
    }
}

// ---------------------------------------------------------------------------
// out = LayerNorm(a + b) * g + beta  (one block per token row); optionally
// also writes the result in packed-A bf16 layout for the next GEMM.
// ---------------------------------------------------------------------------
__global__ __launch_bounds__(THREADS) void k_add_ln(const float* __restrict__ a,
                                                    const float* __restrict__ b,
                                                    const float* __restrict__ g,
                                                    const float* __restrict__ beta,
                                                    float* __restrict__ out, int D,
                                                    __bf16* __restrict__ outp, int packKt) {
    size_t base = (size_t)blockIdx.x * D;
    __shared__ float s1[THREADS], s2[THREADS];
    int tid = threadIdx.x;
    float acc = 0.0f, acc2 = 0.0f;
    for (int i = tid; i < D; i += THREADS) {
        float v = a[base + i] + b[base + i];
        acc += v; acc2 += v * v;
    }
    s1[tid] = acc; s2[tid] = acc2; __syncthreads();
    for (int o = THREADS / 2; o > 0; o >>= 1) {
        if (tid < o) { s1[tid] += s1[tid + o]; s2[tid] += s2[tid + o]; }
        __syncthreads();
    }
    float mean = s1[0] / (float)D;
    float var  = s2[0] / (float)D - mean * mean;
    float rinv = rsqrtf(var + 1e-5f);
    for (int i = tid; i < D; i += THREADS) {
        float v = (a[base + i] + b[base + i] - mean) * rinv * g[i] + beta[i];
        out[base + i] = v;
        if (outp) outp[pa_idx((int)blockIdx.x, i, packKt)] = (__bf16)v;
    }
}

// ---------------------------------------------------------------------------
// Host orchestration
// ---------------------------------------------------------------------------
static inline int grid1(size_t n) { return (int)((n + THREADS - 1) / THREADS); }

extern "C" void kernel_launch(void* const* d_in, const int* in_sizes, int n_in,
                              void* d_out, int out_size, void* d_ws, size_t ws_size,
                              hipStream_t stream) {
    (void)in_sizes; (void)n_in; (void)out_size; (void)ws_size;
    const int Bb = 2, S = SEQ, D = 1024, H = 16, Dh = 64, F = 4096;
    const int T = Bb * S;                         // 4096 tokens

    const float* x   = (const float*)d_in[0];
    const float* wq  = (const float*)d_in[2];  const float* bq = (const float*)d_in[3];
    const float* wk  = (const float*)d_in[4];  const float* bk = (const float*)d_in[5];
    const float* wv  = (const float*)d_in[6];  const float* bv = (const float*)d_in[7];
    const float* wo  = (const float*)d_in[8];  const float* bo = (const float*)d_in[9];
    const float* w1  = (const float*)d_in[10]; const float* b1 = (const float*)d_in[11];
    const float* w2  = (const float*)d_in[12]; const float* b2 = (const float*)d_in[13];
    const float* g1  = (const float*)d_in[14]; const float* be1= (const float*)d_in[15];
    const float* g2  = (const float*)d_in[16]; const float* be2= (const float*)d_in[17];

    char* ws = (char*)d_ws;
    size_t off = 0;
    auto alloc = [&](size_t bytes) -> char* {
        char* p = ws + off;
        off = (off + bytes + 255) & ~(size_t)255;
        return p;
    };
    __bf16* xp   = (__bf16*)alloc((size_t)T * D * 2);
    __bf16* wqp  = (__bf16*)alloc((size_t)D * D * 2);
    __bf16* wkp  = (__bf16*)alloc((size_t)D * D * 2);
    __bf16* wvp  = (__bf16*)alloc((size_t)D * D * 2);
    __bf16* wop  = (__bf16*)alloc((size_t)D * D * 2);
    __bf16* w1p  = (__bf16*)alloc((size_t)D * F * 2);
    __bf16* w2p  = (__bf16*)alloc((size_t)F * D * 2);
    __bf16* qhp  = (__bf16*)alloc((size_t)Bb * H * S * Dh * 2);  // per-(b,h) A-fmt
    __bf16* kthp = (__bf16*)alloc((size_t)Bb * H * S * Dh * 2);  // per-(b,h) B-fmt (K^T)
    __bf16* vhp  = (__bf16*)alloc((size_t)Bb * H * S * Dh * 2);  // per-(b,h) B-fmt
    float*  scb  = (float*) alloc((size_t)S * S * 4);
    __bf16* attp = (__bf16*)alloc((size_t)S * S * 2);
    __bf16* ctxp = (__bf16*)alloc((size_t)T * D * 2);            // packed-A context
    float*  aprj = (float*) alloc((size_t)T * D * 4);
    float*  x1   = (float*) alloc((size_t)T * D * 4);
    __bf16* x1p  = (__bf16*)alloc((size_t)T * D * 2);
    __bf16* ff1p = (__bf16*)alloc((size_t)T * F * 2);            // packed-A relu(ffn1)
    float*  ff2  = (float*) alloc((size_t)T * D * 4);

    // --- pack x and weights to bf16 WMMA fragments ---
    k_pack_a<<<grid1((size_t)T * D), THREADS, 0, stream>>>(x,  xp,  T, D, D);
    k_pack_b<<<grid1((size_t)D * D), THREADS, 0, stream>>>(wq, wqp, D, D, D);
    k_pack_b<<<grid1((size_t)D * D), THREADS, 0, stream>>>(wk, wkp, D, D, D);
    k_pack_b<<<grid1((size_t)D * D), THREADS, 0, stream>>>(wv, wvp, D, D, D);
    k_pack_b<<<grid1((size_t)D * D), THREADS, 0, stream>>>(wo, wop, D, D, D);
    k_pack_b<<<grid1((size_t)D * F), THREADS, 0, stream>>>(w1, w1p, D, F, F);
    k_pack_b<<<grid1((size_t)F * D), THREADS, 0, stream>>>(w2, w2p, F, D, D);

    // --- QKV projections; epilogue writes per-head WMMA fragments directly ---
    dim3 gProj(T / 128, D / 128);
    k_gemm_big<2><<<gProj, THREADS, 0, stream>>>(xp, wqp, D / 32, nullptr, 0,
                                                 qhp,  0, 0, 0, bq, 0);
    k_gemm_big<3><<<gProj, THREADS, 0, stream>>>(xp, wkp, D / 32, nullptr, 0,
                                                 kthp, 0, 0, 0, bk, 0);
    k_gemm_big<4><<<gProj, THREADS, 0, stream>>>(xp, wvp, D / 32, nullptr, 0,
                                                 vhp,  0, 0, 0, bv, 0);

    // --- attention per (batch, head) ---
    dim3 gSc(S / 128, S / 128);
    dim3 gPV(S / 128, Dh / 64);
    for (int b = 0; b < Bb; ++b) {
        for (int h = 0; h < H; ++h) {
            const size_t ho = (size_t)(b * H + h) * (size_t)S * Dh;
            // scores[S,S] = Q_h @ K_h^T (K=64)
            k_gemm_big<0><<<gSc, THREADS, 0, stream>>>(qhp + ho, kthp + ho, Dh / 32,
                                                       scb, S, nullptr, 0, 0, 0,
                                                       nullptr, 0);
            // causal softmax (scale 1/8) -> packed-A bf16 probs
            k_softmax_pack<<<S, THREADS, 0, stream>>>(scb, attp, S, 0.125f);
            // ctx[:,h*64:+64] = P @ V_h (K=S); writes packed-A ctx directly
            k_gemm_small<1><<<gPV, THREADS, 0, stream>>>(attp, vhp + ho, S / 32,
                                                         nullptr, 0, ctxp, D / 32,
                                                         b * S, h * Dh, nullptr, 0);
        }
    }

    // --- output projection + residual + LN1 (dual fp32 + packed output) ---
    k_gemm_big<0><<<gProj, THREADS, 0, stream>>>(ctxp, wop, D / 32, aprj, D,
                                                 nullptr, 0, 0, 0, bo, 0);
    k_add_ln<<<T, THREADS, 0, stream>>>(x, aprj, g1, be1, x1, D, x1p, D / 32);

    // --- FFN: relu(x1@w1+b1) written straight to packed-A, then @w2+b2 ---
    dim3 gF1(T / 128, F / 128);
    k_gemm_big<1><<<gF1, THREADS, 0, stream>>>(x1p, w1p, D / 32, nullptr, 0,
                                               ff1p, F / 32, 0, 0, b1, 1);
    k_gemm_big<0><<<gProj, THREADS, 0, stream>>>(ff1p, w2p, F / 32, ff2, D,
                                                 nullptr, 0, 0, 0, b2, 0);

    // --- residual + LN2 -> output ---
    k_add_ln<<<T, THREADS, 0, stream>>>(x1, ff2, g2, be2, (float*)d_out, D,
                                        nullptr, 0);
}
